// HSMSSD_60730837565759
// MI455X (gfx1250) — matrix operand order repaired and checked by reading, not compiled, not run on hardware
//
#include <hip/hip_runtime.h>
#include <hip/hip_bf16.h>
#include <stdint.h>

// ---------------------------------------------------------------------------
// HSMSSD on MI455X (gfx1250, wave32). All GEMMs run on v_wmma_f32_16x16x32_bf16
// with f32 accumulation; matrix operands pre-converted to bf16, except K5's
// B (f32 x), which is converted to bf16 inside the GEMM inner loop.
// B=16, d_model=384, L=4096 (64x64), state_dim=64, d_inner=768, conv_dim=192
// ---------------------------------------------------------------------------

typedef __attribute__((ext_vector_type(16))) __bf16 v16bf;
typedef __attribute__((ext_vector_type(8)))  __bf16 v8bf;
typedef __attribute__((ext_vector_type(8)))  float  v8f;

static __device__ __forceinline__ unsigned short f2bf_bits(float f) {
    unsigned int u = __builtin_bit_cast(unsigned int, f);
    u += 0x7fffu + ((u >> 16) & 1u);           // round-to-nearest-even
    return (unsigned short)(u >> 16);
}
static __device__ __forceinline__ __bf16 f2bf(float f) {
    return __builtin_bit_cast(__bf16, f2bf_bits(f));
}

// ---------------------------------------------------------------------------
// Generic f32 -> bf16 conversion (weights etc.)
// ---------------------------------------------------------------------------
__global__ void k_cvt_bf16(const float* __restrict__ src,
                           unsigned short* __restrict__ dst, int n) {
    int i = blockIdx.x * 256 + threadIdx.x;
    if (i < n) dst[i] = f2bf_bits(src[i]);
}

// ---------------------------------------------------------------------------
// K0: transpose x (B,384,4096) f32 -> xT (B,4096,384) bf16 (GEMM1 B operand)
// ---------------------------------------------------------------------------
__global__ void k_transpose_x_bf16(const float* __restrict__ x,
                                   unsigned short* __restrict__ xT) {
    __shared__ float tile[32][33];
    const int b  = blockIdx.z;
    const int d0 = blockIdx.y * 32;
    const int l0 = blockIdx.x * 32;
    const int tx = threadIdx.x;      // 0..31
    const int ty = threadIdx.y;      // 0..7
    const float* xb = x + (long long)b * 384 * 4096;
#pragma unroll
    for (int j = 0; j < 4; ++j) {
        int d = ty + j * 8;
        tile[d][tx] = xb[(long long)(d0 + d) * 4096 + (l0 + tx)];
    }
    __syncthreads();
    unsigned short* xTb = xT + (long long)b * 4096 * 384;
#pragma unroll
    for (int j = 0; j < 4; ++j) {
        int l = ty + j * 8;
        xTb[(long long)(l0 + l) * 384 + (d0 + tx)] = f2bf_bits(tile[tx][l]);
    }
}

// ---------------------------------------------------------------------------
// WMMA GEMM, register-blocked: each wave computes MREP vertically-stacked
// 16x16 tiles. Per K-step: one B fragment load, MREP A loads + MREP WMMAs
// (B fragment reused across the stacked tiles).
//   D[b] = A[b] (bf16, row-major MxK) * B[b] (KxN, stored as BT[n][k])
// A fragment: 32B contiguous at K = kk+half*16, row m = lane&15.
// B fragment: 32B contiguous at K = kk+half*16, column n = lane&15.
// Outputs: optional f32 D and/or bf16 Dh (same geometry).
// Tile guard is wave-uniform => EXEC all-ones at every v_wmma.
// ---------------------------------------------------------------------------
template <int MREP, bool B_IS_BF16>
__global__ void k_wmma_gemm(const unsigned short* __restrict__ A,
                            const void*  __restrict__ BT,
                            float* __restrict__ D,
                            unsigned short* __restrict__ Dh,
                            int Mg, int Nt, int K,
                            int lda, int ldb, int ldd,
                            long long sA, long long sB, long long sD,
                            int nBatch) {
    const int wave  = blockIdx.x * (blockDim.x >> 5) + (threadIdx.x >> 5);
    const int total = nBatch * Mg * Nt;
    if (wave >= total) return;                 // wave-uniform exit
    const int bz = wave / (Mg * Nt);
    const int rr = wave - bz * (Mg * Nt);
    const int mg = rr / Nt;
    const int nt = rr - mg * Nt;
    const int lane = threadIdx.x & 31;
    const int lm = lane & 15;                  // row (A) / col (B,D) index
    const int lh = lane >> 4;                  // half select

    // A base: folded lane row + half offset (lh*16)
    const unsigned short* Abase =
        (const unsigned short*)A + bz * sA +
        (long long)(mg * MREP * 16 + lm) * lda + lh * 16;

    v8f acc[MREP];
#pragma unroll
    for (int m = 0; m < MREP; ++m) acc[m] = v8f{};

    if constexpr (B_IS_BF16) {
        const unsigned short* Bb = (const unsigned short*)BT + bz * sB +
                                   (long long)(nt * 16 + lm) * ldb + lh * 16;
        for (int kk = 0; kk < K; kk += 32) {
            const v16bf bfrag = *(const v16bf*)(Bb + kk);
#pragma unroll
            for (int m = 0; m < MREP; ++m) {
                const unsigned short* pa = Abase + (long long)m * 16 * lda + kk;
                const v8bf lo = *(const v8bf*)(pa);
                const v8bf hi = *(const v8bf*)(pa + 8);
                v16bf a;
#pragma unroll
                for (int i = 0; i < 8; ++i) { a[i] = lo[i]; a[8 + i] = hi[i]; }
                acc[m] = __builtin_amdgcn_wmma_f32_16x16x32_bf16(
                             false, a, false, bfrag, (short)0, acc[m], false, false);
            }
        }
    } else {
        const float* Bb = (const float*)BT + bz * sB +
                          (long long)(nt * 16 + lm) * ldb + lh * 16;
        for (int kk = 0; kk < K; kk += 32) {
            const float* pb = Bb + kk;
            v16bf bfrag;
#pragma unroll
            for (int i = 0; i < 16; ++i) bfrag[i] = f2bf(pb[i]);
#pragma unroll
            for (int m = 0; m < MREP; ++m) {
                const unsigned short* pa = Abase + (long long)m * 16 * lda + kk;
                const v8bf lo = *(const v8bf*)(pa);
                const v8bf hi = *(const v8bf*)(pa + 8);
                v16bf a;
#pragma unroll
                for (int i = 0; i < 8; ++i) { a[i] = lo[i]; a[8 + i] = hi[i]; }
                acc[m] = __builtin_amdgcn_wmma_f32_16x16x32_bf16(
                             false, a, false, bfrag, (short)0, acc[m], false, false);
            }
        }
    }

    // D layout (ISA 7.12.2): VGPR r -> row r + 8*half, col = lane&15
#pragma unroll
    for (int m = 0; m < MREP; ++m) {
        const long long rowBase = (long long)((mg * MREP + m) * 16);
        if (D != nullptr) {
            float* Db = D + bz * sD + rowBase * ldd + nt * 16;
#pragma unroll
            for (int r = 0; r < 8; ++r)
                Db[(long long)(r + 8 * lh) * ldd + lm] = acc[m][r];
        }
        if (Dh != nullptr) {
            unsigned short* Dhb = Dh + bz * sD + rowBase * ldd + nt * 16;
#pragma unroll
            for (int r = 0; r < 8; ++r)
                Dhb[(long long)(r + 8 * lh) * ldd + lm] = f2bf_bits(acc[m][r]);
        }
    }
}

// ---------------------------------------------------------------------------
// K2: depthwise 3x3 conv, pad 1, on (B,192,64,64) f32
// ---------------------------------------------------------------------------
__global__ void k_dwconv3x3(const float* __restrict__ in,
                            const float* __restrict__ w,
                            float* __restrict__ out) {
    const int idx = blockIdx.x * 256 + threadIdx.x;   // < 16*192*4096
    const int xc = idx & 63;
    const int yr = (idx >> 6) & 63;
    const int plane = idx >> 12;                      // b*192 + c
    const int c = plane % 192;
    const float* p = in + (long long)plane * 4096;
    const float* wc = w + c * 9;
    float acc = 0.f;
#pragma unroll
    for (int ky = 0; ky < 3; ++ky) {
        const int yy = yr + ky - 1;
        if ((unsigned)yy < 64u) {
#pragma unroll
            for (int kx = 0; kx < 3; ++kx) {
                const int xx = xc + kx - 1;
                if ((unsigned)xx < 64u)
                    acc += wc[ky * 3 + kx] * p[yy * 64 + xx];
            }
        }
    }
    out[idx] = acc;
}

// ---------------------------------------------------------------------------
// K3: per-(b,s) softmax over L=4096 of dt, then AB = softmax(dt)*Bm -> bf16.
// (a_param is constant along the softmax axis so it cancels exactly.)
// ---------------------------------------------------------------------------
__global__ void k_softmax_ab(const float* __restrict__ conv,
                             unsigned short* __restrict__ AB) {
    const int row = blockIdx.x;              // 0..1023; b = row>>6, s = row&63
    const int b = row >> 6, s = row & 63;
    const float* dt = conv + (long long)(b * 192 + 128 + s) * 4096;
    const float* bm = conv + (long long)(b * 192 + s) * 4096;
    unsigned short* ab = AB + (long long)row * 4096;
    const int t = threadIdx.x;
    __shared__ float red[8];

    float v[16];
    float mx = -3.0e38f;
#pragma unroll
    for (int i = 0; i < 16; ++i) { v[i] = dt[t + 256 * i]; mx = fmaxf(mx, v[i]); }
#pragma unroll
    for (int o = 16; o > 0; o >>= 1) mx = fmaxf(mx, __shfl_xor(mx, o, 32));
    if ((t & 31) == 0) red[t >> 5] = mx;
    __syncthreads();
    mx = red[0];
#pragma unroll
    for (int wv = 1; wv < 8; ++wv) mx = fmaxf(mx, red[wv]);

    float sum = 0.f;
#pragma unroll
    for (int i = 0; i < 16; ++i) { v[i] = __expf(v[i] - mx); sum += v[i]; }
#pragma unroll
    for (int o = 16; o > 0; o >>= 1) sum += __shfl_xor(sum, o, 32);
    __syncthreads();
    if ((t & 31) == 0) red[t >> 5] = sum;
    __syncthreads();
    sum = 0.f;
#pragma unroll
    for (int wv = 0; wv < 8; ++wv) sum += red[wv];
    const float inv = 1.f / sum;
#pragma unroll
    for (int i = 0; i < 16; ++i)
        ab[t + 256 * i] = f2bf_bits(v[i] * inv * bm[t + 256 * i]);
}

// ---------------------------------------------------------------------------
// K4: CmT (B,4096,64) bf16 <- transpose of conv channels [64,128)
// ---------------------------------------------------------------------------
__global__ void k_transpose_cm(const float* __restrict__ conv,
                               unsigned short* __restrict__ CmT) {
    __shared__ float tile[32][33];
    const int b  = blockIdx.z;
    const int s0 = blockIdx.y * 32;
    const int l0 = blockIdx.x * 32;
    const int tx = threadIdx.x, ty = threadIdx.y;
#pragma unroll
    for (int j = 0; j < 4; ++j) {
        int s = ty + j * 8;
        tile[s][tx] = conv[(long long)(b * 192 + 64 + s0 + s) * 4096 + (l0 + tx)];
    }
    __syncthreads();
#pragma unroll
    for (int j = 0; j < 4; ++j) {
        int l = ty + j * 8;
        CmT[(long long)(b * 4096 + l0 + l) * 64 + (s0 + tx)] = f2bf_bits(tile[tx][l]);
    }
}

// ---------------------------------------------------------------------------
// K7: gate g = h_i*silu(z) + h_i*d ; hz f32 (B,1536,64) -> gT bf16 (B,64,768)
// ---------------------------------------------------------------------------
__global__ void k_gate(const float* __restrict__ hz,
                       const float* __restrict__ dpar,
                       unsigned short* __restrict__ gT) {
    const int idx = blockIdx.x * 256 + threadIdx.x;
    if (idx >= 16 * 768 * 64) return;
    const int o = idx % 768;
    const int t = idx / 768;
    const int s = t & 63;
    const int b = t >> 6;
    const float hi = hz[(long long)(b * 1536 + o) * 64 + s];
    const float z  = hz[(long long)(b * 1536 + 768 + o) * 64 + s];
    const float sil = z * (1.f / (1.f + __expf(-z)));
    gT[(long long)(b * 64 + s) * 768 + o] = f2bf_bits(hi * (sil + dpar[0]));
}

// ---------------------------------------------------------------------------
// Host launcher
// ---------------------------------------------------------------------------
extern "C" void kernel_launch(void* const* d_in, const int* in_sizes, int n_in,
                              void* d_out, int out_size, void* d_ws, size_t ws_size,
                              hipStream_t stream) {
    (void)in_sizes; (void)n_in; (void)out_size; (void)ws_size;

    const float* x       = (const float*)d_in[0];   // (16,384,4096)
    const float* w_bcdt  = (const float*)d_in[1];   // (192,384)
    const float* w_dw    = (const float*)d_in[2];   // (192,1,3,3)
    const float* w_hz    = (const float*)d_in[3];   // (1536,384)
    const float* w_out   = (const float*)d_in[4];   // (384,768)
    /* d_in[5] = a_param: constant along softmax axis -> cancels; unused */
    const float* d_param = (const float*)d_in[6];   // (1,)
    /* d_in[7], d_in[8] = H, W (== 64, hardcoded) */

    // Workspace layout (bytes). The xT region (dead after GEMM1) is recycled
    // as the pool for all post-conv intermediates.
    char* ws = (char*)d_ws;
    unsigned short* xT     = (unsigned short*)(ws + 0);        // 50.3 MB bf16
    unsigned short* AB     = (unsigned short*)(ws + 0);        //  8.4 MB bf16
    unsigned short* CmT    = (unsigned short*)(ws + 8388608);  //  8.4 MB bf16
    unsigned short* hT     = (unsigned short*)(ws + 16777216); //  0.8 MB bf16
    float*          hz     = (float*)         (ws + 17563648); //  6.3 MB f32
    unsigned short* gT     = (unsigned short*)(ws + 23855104); //  1.6 MB bf16
    unsigned short* houtH  = (unsigned short*)(ws + 25427968); //  0.8 MB bf16
    float*          bcdt   = (float*)         (ws + 50331648); // 50.3 MB f32
    float*          conv   = (float*)         (ws + 100663296);// 50.3 MB f32
    unsigned short* wbcdtH = (unsigned short*)(ws + 150994944);// 147 KB
    unsigned short* whzH   = (unsigned short*)(ws + 151142400);//  1.2 MB
    unsigned short* woutH  = (unsigned short*)(ws + 152322048);//  0.6 MB

    float* y      = (float*)d_out;                 // (16,384,4096)
    float* h_out2 = (float*)d_out + 25165824;      // (16,384,64)

    dim3 tb(32, 8);

    // Weight conversions (bf16, row-major = WMMA A operands)
    k_cvt_bf16<<<(73728  + 255) / 256, 256, 0, stream>>>(w_bcdt, wbcdtH, 73728);
    k_cvt_bf16<<<(589824 + 255) / 256, 256, 0, stream>>>(w_hz,   whzH,   589824);
    k_cvt_bf16<<<(294912 + 255) / 256, 256, 0, stream>>>(w_out,  woutH,  294912);

    // K0: x -> xT bf16
    k_transpose_x_bf16<<<dim3(128, 12, 16), tb, 0, stream>>>(x, xT);

    // K1: bc_dt[b] (192x4096) = w_bcdt (192x384) * x[b] (384x4096)
    {
        int Mg = 3, Nt = 256, total = 16 * Mg * Nt;   // MREP=4 -> 12 M-tiles
        k_wmma_gemm<4, true><<<(total + 7) / 8, 256, 0, stream>>>(
            wbcdtH, xT, bcdt, nullptr, Mg, Nt, 384,
            384, 384, 4096, 0LL, 4096LL * 384, 192LL * 4096, 16);
    }

    // K2: depthwise conv
    k_dwconv3x3<<<(16 * 192 * 4096) / 256, 256, 0, stream>>>(bcdt, w_dw, conv);

    // K3: AB = softmax(dt) * Bm  (bf16 out)
    k_softmax_ab<<<1024, 256, 0, stream>>>(conv, AB);

    // K4: CmT bf16
    k_transpose_cm<<<dim3(128, 2, 16), tb, 0, stream>>>(conv, CmT);

    // K5: hT[b] (64x384) = AB[b] (64x4096) * x[b]^T  (B-frags read f32 x)
    {
        int Mg = 1, Nt = 24, total = 16 * Mg * Nt;
        k_wmma_gemm<4, false><<<(total + 7) / 8, 256, 0, stream>>>(
            AB, x, nullptr, hT, Mg, Nt, 4096,
            4096, 4096, 384, 64LL * 4096, 384LL * 4096, 64LL * 384, 16);
    }

    // K6: hz[b] (1536x64) = w_hz (1536x384) * h[b] (384x64), B from hT
    {
        int Mg = 24, Nt = 4, total = 16 * Mg * Nt;
        k_wmma_gemm<4, true><<<(total + 7) / 8, 256, 0, stream>>>(
            whzH, hT, hz, nullptr, Mg, Nt, 384,
            384, 384, 64, 0LL, 64LL * 384, 1536LL * 64, 16);
    }

    // K7: gate -> gT bf16
    k_gate<<<(16 * 768 * 64 + 255) / 256, 256, 0, stream>>>(hz, d_param, gT);

    // K8: hout[b] (384x64) = w_out (384x768) * g[b] (768x64)
    //     f32 copy -> d_out (h output), bf16 copy -> GEMM9 A operand
    {
        int Mg = 6, Nt = 4, total = 16 * Mg * Nt;
        k_wmma_gemm<4, true><<<(total + 7) / 8, 256, 0, stream>>>(
            woutH, gT, h_out2, houtH, Mg, Nt, 768,
            768, 768, 64, 0LL, 64LL * 768, 384LL * 64, 16);
    }

    // K9: y[b] (384x4096) = hout[b] (384x64) * Cm[b] (64x4096), B from CmT
    {
        int Mg = 6, Nt = 256, total = 16 * Mg * Nt;
        k_wmma_gemm<4, true><<<(total + 7) / 8, 256, 0, stream>>>(
            houtH, CmT, y, nullptr, Mg, Nt, 64,
            64, 64, 4096, 384LL * 64, 4096LL * 64, 384LL * 4096, 16);
    }
}
